// Model_46978352283874
// MI455X (gfx1250) — matrix operand order repaired and checked by reading, not compiled, not run on hardware
//
#include <hip/hip_runtime.h>
#include <hip/hip_bf16.h>

typedef __attribute__((ext_vector_type(16))) _Float16 v16h;
typedef __attribute__((ext_vector_type(8)))  float    v8f;

#define LTOK   512          // L*B
#define ND     1024
#define NE     256
#define NV     128000
#define C0     10000
#define TAILN  118000       // NV - C0
#define NTILES 7375         // TAILN/16
#define GT     16           // col tiles per tail block
#define NGRP   461          // ceil(7375/16)
#define HCOLS  10016        // padded head logit row
#define NEGINF (-1e30f)

// --- fragment loaders (contiguous k order, same for A and B) --------------
// A 16x32: lane L holds row L%16; elem e -> K=16*(L/16)+e
// B 32x16: lane L holds col L%16; elem e -> K=16*(L/16)+e
__device__ inline void pack8(v16h& d, int o, float4 x, float4 y) {
    d[o + 0] = (_Float16)x.x; d[o + 1] = (_Float16)x.y;
    d[o + 2] = (_Float16)x.z; d[o + 3] = (_Float16)x.w;
    d[o + 4] = (_Float16)y.x; d[o + 5] = (_Float16)y.y;
    d[o + 6] = (_Float16)y.z; d[o + 7] = (_Float16)y.w;
}
__device__ inline v16h load_afrag(const float* __restrict__ Ab, int kk, int half) {
    v16h a;
    const float4* p0 = (const float4*)(Ab + kk + 16 * half);
    pack8(a, 0, p0[0], p0[1]);
    pack8(a, 8, p0[2], p0[3]);
    return a;
}
__device__ inline v16h load_bfrag_row(const float* __restrict__ Bn, int kk, int half) {
    v16h b;
    const float4* p = (const float4*)(Bn + kk + 16 * half);
    pack8(b, 0, p[0], p[1]);
    pack8(b, 8, p[2], p[3]);
    return b;
}

// ---------------------------------------------------------------------------
// Generic f32-in/f32-out GEMM via v_wmma_f32_16x16x32_f16, branch-free inner loop.
// C[m,n] = alpha * sum_k A[g(m),k]*B(k,n) + bias[n]
// grid = (N/16, M/16), block = 32 (one wave per 16x16 tile).
// Invalid columns (CHECKN) load row 0 of B; their lane-private results are
// replaced by `fill` at store time (B lane data only feeds its own column).
// ---------------------------------------------------------------------------
template <int TRANSB, int CHECKN, int GATHER>
__global__ __launch_bounds__(32) void gemm_wmma(
    const float* __restrict__ A, int lda, const int* __restrict__ gather,
    const float* __restrict__ B, int ldb,
    float* __restrict__ C, int ldc, int K, int Nreal,
    float alpha, const float* __restrict__ bias, float fill)
{
    int lane = threadIdx.x & 31;
    int half = lane >> 4, lm = lane & 15;
    int m = blockIdx.y * 16 + lm;
    int n = blockIdx.x * 16 + lm;
    bool nvalid = CHECKN ? (n < Nreal) : true;
    int nsafe = nvalid ? n : 0;
    int arow = GATHER ? gather[m] : m;
    const float* Ab = A + (size_t)arow * lda;
    const float* Bn = B + (size_t)nsafe * ldb;   // TRANSB path

    v8f acc = {};
    for (int kk = 0; kk < K; kk += 32) {
        v16h a = load_afrag(Ab, kk, half);
        v16h b;
        if (TRANSB) {
            b = load_bfrag_row(Bn, kk, half);
        } else {
#pragma unroll
            for (int e = 0; e < 16; ++e)
                b[e] = (_Float16)B[(size_t)(kk + 16 * half + e) * ldb + n];
        }
        acc = __builtin_amdgcn_wmma_f32_16x16x32_f16(false, a, false, b,
                                                     (short)0, acc, false, false);
    }
    float badd = bias ? bias[nsafe] : 0.f;
#pragma unroll
    for (int r = 0; r < 8; ++r) {
        int mm = blockIdx.y * 16 + r + 8 * half;
        C[(size_t)mm * ldc + n] = nvalid ? alpha * acc[r] + badd : fill;
    }
}

// ---------------------------------------------------------------------------
// SRU elementwise recurrence: one thread per (b,d) channel, sequential over t.
// ---------------------------------------------------------------------------
__global__ void sru_scan(const float* __restrict__ U, float* __restrict__ h,
                         const float* __restrict__ c0, const float* __restrict__ wc,
                         const float* __restrict__ bvec, float* __restrict__ cout)
{
    int idx = blockIdx.x * blockDim.x + threadIdx.x;   // b*1024 + d
    if (idx >= 4096) return;
    int bcol = idx >> 10;
    int d = idx & 1023;
    float vf = wc[d], vr = wc[ND + d];
    float bf = bvec[d] - 3.0f;           // HIGHWAY_BIAS
    float br = bvec[ND + d];
    float c = c0[idx];
    for (int t = 0; t < 128; ++t) {
        int m = t * 4 + bcol;
        const float* um = U + (size_t)m * 3072;
        float u0 = um[d], u1 = um[ND + d], u2 = um[2 * ND + d];
        float xt = h[(size_t)m * ND + d];
        float f = 1.f / (1.f + __expf(-(u1 + vf * c + bf)));
        float r = 1.f / (1.f + __expf(-(u2 + vr * c + br)));
        c = f * c + (1.f - f) * u0;
        h[(size_t)m * ND + d] = r * c + (1.f - r) * xt;
    }
    cout[idx] = c;
}

// ---------------------------------------------------------------------------
// Per-row max + log-sum-exp over padded head logits (pads are -1e30 -> exp=0).
// ---------------------------------------------------------------------------
__global__ void head_reduce(const float* __restrict__ logits,
                            float* __restrict__ hm, float* __restrict__ hls)
{
    __shared__ float red[256];
    int row = blockIdx.x, tid = threadIdx.x;
    const float* lrow = logits + (size_t)row * HCOLS;
    float m = NEGINF;
    for (int c = tid; c < HCOLS; c += 256) m = fmaxf(m, lrow[c]);
    red[tid] = m; __syncthreads();
    for (int s = 128; s > 0; s >>= 1) {
        if (tid < s) red[tid] = fmaxf(red[tid], red[tid + s]);
        __syncthreads();
    }
    m = red[0]; __syncthreads();
    float sum = 0.f;
    for (int c = tid; c < HCOLS; c += 256) sum += __expf(lrow[c] - m);
    red[tid] = sum; __syncthreads();
    for (int s = 128; s > 0; s >>= 1) {
        if (tid < s) red[tid] += red[tid + s];
        __syncthreads();
    }
    if (tid == 0) { hm[row] = m; hls[row] = logf(red[0]); }
}

// ---------------------------------------------------------------------------
// Tail streaming logsumexp. Block = 4 waves, owns GT=16 column tiles (256 cols):
// emb_weight tail streamed once across the grid. B fragments for a column tile
// are preloaded into 64 VGPRs and reused across all 8 row tiles of the wave.
// Wave w owns rows [128w,128w+128) -> race-free per-(row,seg) LDS accumulators.
// Cluster boundaries are multiples of 16, so each 16-col tile is one segment.
// ---------------------------------------------------------------------------
__global__ __launch_bounds__(128) void tail_lse_kernel(
    const float* __restrict__ proj, const float* __restrict__ embw,
    const float* __restrict__ obias, float* __restrict__ part)
{
    __shared__ float accM[LTOK * 5];
    __shared__ float accS[LTOK * 5];
    int tid = threadIdx.x;
    for (int i = tid; i < LTOK * 5; i += 128) { accM[i] = NEGINF; accS[i] = 0.f; }
    __syncthreads();

    int wave = tid >> 5, lane = tid & 31, half = lane >> 4, lm = lane & 15;
    int g = blockIdx.x;

    for (int ti = 0; ti < GT; ++ti) {
        int ct = g * GT + ti;
        if (ct >= NTILES) break;
        int cbase = ct * 16;
        int seg = (cbase >= 90000) ? 4 : (cbase >= 50000) ? 3
                : (cbase >= 30000) ? 2 : (cbase >= 10000) ? 1 : 0;
        int col = cbase + lm;
        float bb = obias[C0 + col];
        const float* Brow = embw + (size_t)(C0 + col) * NE;

        v16h bfr[8];
#pragma unroll
        for (int ki = 0; ki < 8; ++ki)
            bfr[ki] = load_bfrag_row(Brow, ki * 32, half);

        for (int rt = 0; rt < 8; ++rt) {
            int rbase = wave * 128 + rt * 16;
            const float* Arow = proj + (size_t)(rbase + lm) * NE;
            v8f acc = {};
#pragma unroll
            for (int ki = 0; ki < 8; ++ki) {
                v16h a = load_afrag(Arow, ki * 32, half);
                acc = __builtin_amdgcn_wmma_f32_16x16x32_f16(false, a, false, bfr[ki],
                                                             (short)0, acc, false, false);
            }
            // per-row logsumexp over the 16 columns of this tile
            float mr[8], sr[8];
#pragma unroll
            for (int r = 0; r < 8; ++r) mr[r] = acc[r] + bb;
#pragma unroll
            for (int off = 1; off < 16; off <<= 1)
#pragma unroll
                for (int r = 0; r < 8; ++r)
                    mr[r] = fmaxf(mr[r], __shfl_xor(mr[r], off, 32));
#pragma unroll
            for (int r = 0; r < 8; ++r) sr[r] = __expf(acc[r] + bb - mr[r]);
#pragma unroll
            for (int off = 1; off < 16; off <<= 1)
#pragma unroll
                for (int r = 0; r < 8; ++r)
                    sr[r] += __shfl_xor(sr[r], off, 32);
            if (lm == 0) {
#pragma unroll
                for (int r = 0; r < 8; ++r) {
                    int row = rbase + 8 * half + r;
                    int idx = row * 5 + seg;
                    float M = accM[idx], S = accS[idx];
                    float m2 = mr[r], s2 = sr[r];
                    if (m2 > M) { accS[idx] = S * __expf(M - m2) + s2; accM[idx] = m2; }
                    else        { accS[idx] = S + s2 * __expf(m2 - M); }
                }
            }
        }
    }
    __syncthreads();
    float* pg = part + (size_t)g * (LTOK * 5 * 2);
    for (int i = tid; i < LTOK * 5; i += 128) { pg[2 * i] = accM[i]; pg[2 * i + 1] = accS[i]; }
}

__global__ void tail_combine(const float* __restrict__ part, float* __restrict__ lse)
{
    int e = blockIdx.x * blockDim.x + threadIdx.x;   // row*5 + seg
    if (e >= LTOK * 5) return;
    float M = NEGINF, S = 0.f;
    for (int g = 0; g < NGRP; ++g) {
        float m2 = part[(size_t)g * (LTOK * 5 * 2) + 2 * e];
        float s2 = part[(size_t)g * (LTOK * 5 * 2) + 2 * e + 1];
        if (m2 > M) { S = S * __expf(M - m2) + s2; M = m2; }
        else        { S += s2 * __expf(m2 - M); }
    }
    lse[e] = M + logf(S);
}

// logit_y[n] = dot(proj_h[n], emb_weight[y[n]]) + out_bias[y[n]]
__global__ void logit_y_kernel(const float* __restrict__ proj, const float* __restrict__ embw,
                               const float* __restrict__ obias, const int* __restrict__ y,
                               float* __restrict__ out)
{
    __shared__ float red[256];
    int n = blockIdx.x, k = threadIdx.x;
    int yf = y[n];
    red[k] = proj[(size_t)n * NE + k] * embw[(size_t)yf * NE + k];
    __syncthreads();
    for (int s = 128; s > 0; s >>= 1) {
        if (k < s) red[k] += red[k + s];
        __syncthreads();
    }
    if (k == 0) out[n] = red[0] + obias[yf];
}

__global__ void final_loss(const float* __restrict__ logits, const float* __restrict__ hm,
                           const float* __restrict__ hls, const float* __restrict__ lse,
                           const float* __restrict__ logy, const int* __restrict__ y,
                           float* __restrict__ loss)
{
    int n = blockIdx.x * blockDim.x + threadIdx.x;
    if (n >= LTOK) return;
    int yf = y[n];
    int cluster = (yf >= 10000) + (yf >= 20000) + (yf >= 40000) + (yf >= 60000) + (yf >= 100000);
    int tci = cluster - 1; if (tci < 0) tci = 0; if (tci > 4) tci = 4;
    float D = hm[n] + hls[n];
    int hcol = yf < C0 ? yf : C0 - 1;
    float head_tok_lp = logits[(size_t)n * HCOLS + hcol] - D;
    float cluster_lp  = logits[(size_t)n * HCOLS + C0 + tci] - D;
    float tail_lp_y   = logy[n] - lse[n * 5 + tci];
    float lp = (cluster == 0) ? head_tok_lp : (cluster_lp + tail_lp_y);
    loss[n] = -lp;
}

// ---------------------------------------------------------------------------
extern "C" void kernel_launch(void* const* d_in, const int* in_sizes, int n_in,
                              void* d_out, int out_size, void* d_ws, size_t ws_size,
                              hipStream_t stream)
{
    const int*   x      = (const int*)d_in[0];
    const int*   y      = (const int*)d_in[1];
    const float* hidden = (const float*)d_in[2];
    const float* embw   = (const float*)d_in[3];
    const float* embp   = (const float*)d_in[4];
    const float* obias  = (const float*)d_in[5];
    const float* cw     = (const float*)d_in[6];
    const float* cb     = (const float*)d_in[7];
    const float* Wp     = (const float*)d_in[8];
    const float* Wq     = (const float*)d_in[9];
    const float* wc     = (const float*)d_in[10];
    const float* bv     = (const float*)d_in[11];
    float* out = (float*)d_out;
    float* ws  = (float*)d_ws;

    float* h      = ws;                 // 512*1024
    float* P      = h + 524288;         // 512*256
    float* U      = P + 131072;         // 512*3072
    float* proj   = U + 1572864;        // 512*256
    float* logits = proj + 131072;      // 512*10016
    float* hm     = logits + 5128192;   // 512
    float* hls    = hm + 512;           // 512
    float* part   = hls + 512;          // 461*512*5*2
    float* lse    = part + (size_t)NGRP * LTOK * 5 * 2;  // 2560
    float* logy   = lse + 2560;         // 512

    dim3 wv(32);
    // h0 = emb_weight[x] @ emb_proj^T * sqrt(1024)
    gemm_wmma<1, 0, 1><<<dim3(64, 32), wv, 0, stream>>>(embw, NE, x, embp, NE,
        h, ND, NE, ND, 32.0f, nullptr, 0.f);
    for (int l = 0; l < 4; ++l) {
        gemm_wmma<0, 0, 0><<<dim3(16, 32), wv, 0, stream>>>(h, ND, nullptr,
            Wp + (size_t)l * ND * NE, NE, P, NE, ND, NE, 1.f, nullptr, 0.f);
        gemm_wmma<0, 0, 0><<<dim3(192, 32), wv, 0, stream>>>(P, NE, nullptr,
            Wq + (size_t)l * NE * 3072, 3072, U, 3072, NE, 3072, 1.f, nullptr, 0.f);
        sru_scan<<<16, 256, 0, stream>>>(U, h, hidden + l * 4096,
                                         wc + l * 2048, bv + l * 2048,
                                         out + 512 + l * 4096);
    }
    // proj_h = h @ emb_proj
    gemm_wmma<0, 0, 0><<<dim3(16, 32), wv, 0, stream>>>(h, ND, nullptr, embp, NE,
        proj, NE, ND, NE, 1.f, nullptr, 0.f);
    // head logits: [emb_weight[:10000] | cluster_weight] (+ biases), padded to 10016
    gemm_wmma<1, 0, 0><<<dim3(625, 32), wv, 0, stream>>>(proj, NE, nullptr, embw, NE,
        logits, HCOLS, NE, C0, 1.f, obias, 0.f);
    gemm_wmma<1, 1, 0><<<dim3(1, 32), wv, 0, stream>>>(proj, NE, nullptr, cw, NE,
        logits + C0, HCOLS, NE, 5, 1.f, cb, NEGINF);
    head_reduce<<<512, 256, 0, stream>>>(logits, hm, hls);
    tail_lse_kernel<<<NGRP, 128, 0, stream>>>(proj, embw, obias, part);
    tail_combine<<<10, 256, 0, stream>>>(part, lse);
    logit_y_kernel<<<512, 256, 0, stream>>>(proj, embw, obias, y, logy);
    final_loss<<<2, 256, 0, stream>>>(logits, hm, hls, lse, logy, y, out);
}
